// SparseAttention_25039659336136
// MI455X (gfx1250) — matrix operand order repaired and checked by reading, not compiled, not run on hardware
//
#include <hip/hip_runtime.h>
#include <stdint.h>

// Problem constants (from reference)
#define BB 2
#define SS 2048
#define DD 1024
#define HH 16
#define DKK 64
#define NROW (BB * SS)            // 4096
#define HALFW 32                  // LW/2

typedef __attribute__((ext_vector_type(16))) __bf16 v16bf;
typedef __attribute__((ext_vector_type(8)))  float  v8f;
typedef __attribute__((ext_vector_type(8)))  __bf16 bf8;

// ---------- helpers ----------
__device__ __forceinline__ __bf16 f2bf(float f) {
  uint32_t u = __builtin_bit_cast(uint32_t, f);
  uint32_t r = u + 0x7FFFu + ((u >> 16) & 1u);   // round-to-nearest-even
  unsigned short h = (unsigned short)(r >> 16);
  return __builtin_bit_cast(__bf16, h);
}

// A-fragment (16x32 bf16, row-major source, per ISA table):
// lane m = lane&15 ; K = 8*(lane>>4) + {0..7} in elems 0..7, +16 in elems 8..15
__device__ __forceinline__ v16bf load_a_frag(const __bf16* __restrict__ base,
                                             int lda, int m0, int kc, int lane) {
  int m = lane & 15;
  int koff = (lane >> 4) << 3;               // 0 or 8
  const __bf16* p = base + (size_t)(m0 + m) * lda + kc + koff;
  bf8 lo = *(const bf8*)p;
  bf8 hi = *(const bf8*)(p + 16);
  v16bf a;
#pragma unroll
  for (int i = 0; i < 8; ++i) { a[i] = lo[i]; a[i + 8] = hi[i]; }
  return a;
}

// B-fragment (32x16 bf16) where B[k][n] = Wrow_n[k] (contiguous in k):
// lane column n = lane&15 ; K = 8*(lane>>4) + {0..7} in elems 0..7, +16 in elems 8..15
__device__ __forceinline__ v16bf load_b_frag(const __bf16* __restrict__ Wrow,
                                             int kc, int lane) {
  int koff = (lane >> 4) << 3;               // 0 or 8
  const __bf16* p = Wrow + kc + koff;
  bf8 lo = *(const bf8*)p;
  bf8 hi = *(const bf8*)(p + 16);
  v16bf b;
#pragma unroll
  for (int i = 0; i < 8; ++i) { b[i] = lo[i]; b[i + 8] = hi[i]; }
  return b;
}

#define WMMA_BF16(A, B, C) \
  __builtin_amdgcn_wmma_f32_16x16x32_bf16(false, (A), false, (B), (short)0, (C), false, false)

// 32x64 output tile per wave, K = DD. Ping-pong double buffering with the K
// loop unrolled by 2 so the buffer swap needs no register moves.
// acc[0..3] = rows m0..m0+15 x 4 n-tiles ; acc[4..7] = rows m0+16..m0+31.
__device__ __forceinline__ void gemm32x64(const __bf16* __restrict__ X,
                                          const __bf16* __restrict__ W,
                                          int m0, int n0, int lane, v8f* acc) {
  const __bf16* wrow[4];
#pragma unroll
  for (int t = 0; t < 4; ++t)
    wrow[t] = W + (size_t)(n0 + (t << 4) + (lane & 15)) * DD;

  // slot A <- chunk 0
  v16bf a0A = load_a_frag(X, DD, m0, 0, lane);
  v16bf a1A = load_a_frag(X, DD, m0 + 16, 0, lane);
  v16bf bA0 = load_b_frag(wrow[0], 0, lane);
  v16bf bA1 = load_b_frag(wrow[1], 0, lane);
  v16bf bA2 = load_b_frag(wrow[2], 0, lane);
  v16bf bA3 = load_b_frag(wrow[3], 0, lane);

  for (int kc = 0; kc < DD; kc += 64) {
    // slot B <- chunk kc+32 (always in range: DD multiple of 64)
    const int k1 = kc + 32;
    v16bf a0B = load_a_frag(X, DD, m0, k1, lane);
    v16bf a1B = load_a_frag(X, DD, m0 + 16, k1, lane);
    v16bf bB0 = load_b_frag(wrow[0], k1, lane);
    v16bf bB1 = load_b_frag(wrow[1], k1, lane);
    v16bf bB2 = load_b_frag(wrow[2], k1, lane);
    v16bf bB3 = load_b_frag(wrow[3], k1, lane);
    __builtin_prefetch(X + (size_t)(m0 + (lane & 15)) * DD + kc + 512, 0, 3);

    // compute chunk kc from slot A
    acc[0] = WMMA_BF16(a0A, bA0, acc[0]);
    acc[4] = WMMA_BF16(a1A, bA0, acc[4]);
    acc[1] = WMMA_BF16(a0A, bA1, acc[1]);
    acc[5] = WMMA_BF16(a1A, bA1, acc[5]);
    acc[2] = WMMA_BF16(a0A, bA2, acc[2]);
    acc[6] = WMMA_BF16(a1A, bA2, acc[6]);
    acc[3] = WMMA_BF16(a0A, bA3, acc[3]);
    acc[7] = WMMA_BF16(a1A, bA3, acc[7]);

    // slot A <- chunk kc+64 (dummy reload of chunk 0 on the final trip)
    const int k2 = (kc + 64 < DD) ? kc + 64 : 0;
    a0A = load_a_frag(X, DD, m0, k2, lane);
    a1A = load_a_frag(X, DD, m0 + 16, k2, lane);
    bA0 = load_b_frag(wrow[0], k2, lane);
    bA1 = load_b_frag(wrow[1], k2, lane);
    bA2 = load_b_frag(wrow[2], k2, lane);
    bA3 = load_b_frag(wrow[3], k2, lane);

    // compute chunk kc+32 from slot B
    acc[0] = WMMA_BF16(a0B, bB0, acc[0]);
    acc[4] = WMMA_BF16(a1B, bB0, acc[4]);
    acc[1] = WMMA_BF16(a0B, bB1, acc[1]);
    acc[5] = WMMA_BF16(a1B, bB1, acc[5]);
    acc[2] = WMMA_BF16(a0B, bB2, acc[2]);
    acc[6] = WMMA_BF16(a1B, bB2, acc[6]);
    acc[3] = WMMA_BF16(a0B, bB3, acc[3]);
    acc[7] = WMMA_BF16(a1B, bB3, acc[7]);
  }
}

// ---------- kernels ----------

// fp32 -> bf16 convert (grid-stride)
__global__ void la_cvt_bf16(const float* __restrict__ src, __bf16* __restrict__ dst, int n) {
  int i = blockIdx.x * blockDim.x + threadIdx.x;
  int stride = gridDim.x * blockDim.x;
  for (; i < n; i += stride) dst[i] = f2bf(src[i]);
}

// zero-fill (float4 grid-stride) for the attention output region
__global__ void la_fill0(float4* __restrict__ p, size_t n4) {
  size_t i = (size_t)blockIdx.x * blockDim.x + threadIdx.x;
  size_t stride = (size_t)gridDim.x * blockDim.x;
  for (; i < n4; i += stride) p[i] = make_float4(0.f, 0.f, 0.f, 0.f);
}

// QKV projection: Dst[b,h,s,dk] = sum_k X[b*S+s, k] * W[h*64+dk, k]   (x @ W.T)
// One wave computes a 32x64 output tile.
__global__ __launch_bounds__(256) void la_gemm_qkv(
    const __bf16* __restrict__ X,
    const __bf16* __restrict__ Wq, const __bf16* __restrict__ Wk, const __bf16* __restrict__ Wv,
    __bf16* __restrict__ Qh, __bf16* __restrict__ Kh, __bf16* __restrict__ Vh) {
  const int lane = threadIdx.x & 31;
  const int wave = blockIdx.x * (blockDim.x >> 5) + (threadIdx.x >> 5); // 0..2047
  const int m0 = (wave >> 4) << 5;    // 0..4064 step 32
  const int n0 = (wave & 15) << 6;    // 0..960  step 64
  const __bf16* W = (blockIdx.z == 0) ? Wq : (blockIdx.z == 1) ? Wk : Wv;
  __bf16* Dst = (blockIdx.z == 0) ? Qh : (blockIdx.z == 1) ? Kh : Vh;

  v8f acc[8] = {};
  gemm32x64(X, W, m0, n0, lane, acc);

  // C layout: VGPR i -> row = i + 8*(lane>>4), col = lane&15
#pragma unroll
  for (int half = 0; half < 2; ++half) {
#pragma unroll
    for (int t = 0; t < 4; ++t) {
      int col = n0 + (t << 4) + (lane & 15);
      int h = col >> 6, dk = col & 63;
#pragma unroll
      for (int i = 0; i < 8; ++i) {
        int row = m0 + (half << 4) + i + ((lane >> 4) << 3);
        int b = row >> 11, s = row & (SS - 1);
        Dst[(((size_t)(b * HH + h) * SS) + s) * DKK + dk] = f2bf(acc[half * 4 + t][i]);
      }
    }
  }
}

// Windowed attention: per-wave one (b,h,16-query tile). Keys in [q0-32, q0+47].
__global__ __launch_bounds__(256) void la_attn(
    const __bf16* __restrict__ Qh, const __bf16* __restrict__ Kh, const __bf16* __restrict__ Vh,
    __bf16* __restrict__ ctx, float* __restrict__ attn_out) {
  __shared__ float sc_all[8][16][96];
  const int lane = threadIdx.x & 31;
  const int wslot = threadIdx.x >> 5;
  float(*sc)[96] = sc_all[wslot];
  const int wave = blockIdx.x * 8 + wslot;   // 0..4095
  const int qt = wave & 127;                 // S/16 tiles
  const int bh = wave >> 7;                  // b*H + h
  const int q0 = qt << 4;
  const int kbase = q0 - HALFW;              // may be negative
  const __bf16* Qb = Qh + (size_t)bh * SS * DKK;
  const __bf16* Kb = Kh + (size_t)bh * SS * DKK;
  const __bf16* Vb = Vh + (size_t)bh * SS * DKK;

  // ---- scores: S = (Q Kt) * DK^-0.5 over 5 key tiles of 16 (pipelined) ----
  v16bf qa0 = load_a_frag(Qb, DKK, q0, 0, lane);
  v16bf qa1 = load_a_frag(Qb, DKK, q0, 32, lane);

  const __bf16* krow;
  {
    int n_glob = kbase + (lane & 15);
    int nr = n_glob < 0 ? 0 : (n_glob > SS - 1 ? SS - 1 : n_glob);
    krow = Kb + (size_t)nr * DKK;
  }
  v16bf kb0 = load_b_frag(krow, 0, lane);
  v16bf kb1 = load_b_frag(krow, 32, lane);

#pragma unroll
  for (int ct = 0; ct < 5; ++ct) {
    // next tile's B fragments (dummy reload of tile 0 on the last iteration)
    int ctn = (ct + 1 < 5) ? ct + 1 : 0;
    int n_glob = kbase + (ctn << 4) + (lane & 15);
    int nr = n_glob < 0 ? 0 : (n_glob > SS - 1 ? SS - 1 : n_glob);
    const __bf16* nkrow = Kb + (size_t)nr * DKK;
    v16bf nk0 = load_b_frag(nkrow, 0, lane);
    v16bf nk1 = load_b_frag(nkrow, 32, lane);

    v8f s = {};
    s = WMMA_BF16(qa0, kb0, s);
    s = WMMA_BF16(qa1, kb1, s);
    int c = (ct << 4) + (lane & 15);
#pragma unroll
    for (int i = 0; i < 8; ++i) {
      int m = i + ((lane >> 4) << 3);
      int ii = q0 + m, jj = kbase + c;
      bool valid = (jj >= 0) & (jj < SS) & (ii - jj <= HALFW) & (jj - ii <= HALFW);
      sc[m][c] = valid ? s[i] * 0.125f : -3.0e38f;
    }
    kb0 = nk0; kb1 = nk1;
  }
  { // pad columns 80..95 (K-dim padding for P*V)
    int c = 80 + (lane & 15);
#pragma unroll
    for (int i = 0; i < 8; ++i) sc[i + ((lane >> 4) << 3)][c] = -3.0e38f;
  }
  __syncthreads();

  // ---- softmax per row (lanes 0..15 each own one row of 96) ----
  if (lane < 16) {
    const int m = lane;
    float mx = -3.0e38f;
    for (int c = 0; c < 96; ++c) mx = fmaxf(mx, sc[m][c]);
    float sum = 0.f;
    for (int c = 0; c < 96; ++c) { float e = __expf(sc[m][c] - mx); sc[m][c] = e; sum += e; }
    float inv = 1.f / sum;
    int ii = q0 + m;
    float* arow = attn_out + ((size_t)bh * SS + ii) * SS;
    for (int c = 0; c < 96; ++c) {
      float p = sc[m][c] * inv;
      sc[m][c] = p;
      int jj = kbase + c;
      if (jj >= 0 && jj < SS && (ii - jj) <= HALFW && (jj - ii) <= HALFW) arow[jj] = p;
    }
  }
  __syncthreads();

  // ---- O = P @ V  (K-dim 96, N = 64) ----
  v8f o[4] = {};
#pragma unroll
  for (int kc = 0; kc < 96; kc += 32) {
    v16bf a;
    {
      int m = lane & 15;
      int koff = (lane >> 4) << 3;
#pragma unroll
      for (int i = 0; i < 8; ++i) {
        a[i]     = f2bf(sc[m][kc + koff + i]);
        a[i + 8] = f2bf(sc[m][kc + koff + 16 + i]);
      }
    }
#pragma unroll
    for (int t = 0; t < 4; ++t) {
      v16bf bf;
      int n = (t << 4) + (lane & 15);
      int kofB = (lane >> 4) << 3;
#pragma unroll
      for (int k2 = 0; k2 < 16; ++k2) {
        int r = kbase + kc + kofB + k2 + (k2 >= 8 ? 8 : 0);
        r = r < 0 ? 0 : (r > SS - 1 ? SS - 1 : r);   // attn==0 there -> contributes 0
        bf[k2] = Vb[(size_t)r * DKK + n];
      }
      o[t] = WMMA_BF16(a, bf, o[t]);
    }
  }
  // ctx[b*S+s, h*64+dk]
  const int b = bh >> 4, h = bh & 15;
#pragma unroll
  for (int t = 0; t < 4; ++t) {
    int dk = (t << 4) + (lane & 15);
#pragma unroll
    for (int i = 0; i < 8; ++i) {
      int m = i + ((lane >> 4) << 3);
      int s = q0 + m;
      ctx[((size_t)(b * SS + s)) * DD + (h << 6) + dk] = f2bf(o[t][i]);
    }
  }
}

// Output projection: out[m, j] = sum_k ctx[m,k] * Wo[j,k] + bo[j]  (fp32 out)
__global__ __launch_bounds__(256) void la_gemm_out(
    const __bf16* __restrict__ Xc, const __bf16* __restrict__ Wo,
    const float* __restrict__ bo, float* __restrict__ out) {
  const int lane = threadIdx.x & 31;
  const int wave = blockIdx.x * (blockDim.x >> 5) + (threadIdx.x >> 5);
  const int m0 = (wave >> 4) << 5;
  const int n0 = (wave & 15) << 6;

  v8f acc[8] = {};
  gemm32x64(Xc, Wo, m0, n0, lane, acc);

#pragma unroll
  for (int half = 0; half < 2; ++half) {
#pragma unroll
    for (int t = 0; t < 4; ++t) {
      int col = n0 + (t << 4) + (lane & 15);
      float bias = bo[col];
#pragma unroll
      for (int i = 0; i < 8; ++i) {
        int row = m0 + (half << 4) + i + ((lane >> 4) << 3);
        out[(size_t)row * DD + col] = acc[half * 4 + t][i] + bias;
      }
    }
  }
}

// ---------- launcher ----------
extern "C" void kernel_launch(void* const* d_in, const int* in_sizes, int n_in,
                              void* d_out, int out_size, void* d_ws, size_t ws_size,
                              hipStream_t stream) {
  const float* q  = (const float*)d_in[0];
  const float* wq = (const float*)d_in[1];
  const float* wk = (const float*)d_in[2];
  const float* wv = (const float*)d_in[3];
  const float* wo = (const float*)d_in[4];
  const float* bo = (const float*)d_in[5];

  float* out  = (float*)d_out;                          // [B,S,D] fp32
  float* attn = out + (size_t)BB * SS * DD;             // [B,H,S,S] fp32

  // workspace carve-up (bf16), total 48 MiB
  __bf16* qb  = (__bf16*)d_ws;                          // [4096,1024]
  __bf16* wqb = qb  + (size_t)NROW * DD;
  __bf16* wkb = wqb + (size_t)DD * DD;
  __bf16* wvb = wkb + (size_t)DD * DD;
  __bf16* wob = wvb + (size_t)DD * DD;
  __bf16* Qh  = wob + (size_t)DD * DD;                  // [B,H,S,DK]
  __bf16* Kh  = Qh  + (size_t)BB * HH * SS * DKK;
  __bf16* Vh  = Kh  + (size_t)BB * HH * SS * DKK;
  __bf16* ctx = Vh  + (size_t)BB * HH * SS * DKK;       // [4096,1024]

  la_cvt_bf16<<<2048, 256, 0, stream>>>(q,  qb,  NROW * DD);
  la_cvt_bf16<<<1024, 256, 0, stream>>>(wq, wqb, DD * DD);
  la_cvt_bf16<<<1024, 256, 0, stream>>>(wk, wkb, DD * DD);
  la_cvt_bf16<<<1024, 256, 0, stream>>>(wv, wvb, DD * DD);
  la_cvt_bf16<<<1024, 256, 0, stream>>>(wo, wob, DD * DD);

  // 2048 waves per matrix: 128 m-tiles (32 rows) x 16 n-strips (64 cols)
  la_gemm_qkv<<<dim3(256, 1, 3), 256, 0, stream>>>(qb, wqb, wkb, wvb, Qh, Kh, Vh);

  la_fill0<<<4096, 256, 0, stream>>>((float4*)attn,
                                     ((size_t)BB * HH * SS * SS) / 4);

  la_attn<<<512, 256, 0, stream>>>(Qh, Kh, Vh, ctx, attn);

  la_gemm_out<<<256, 256, 0, stream>>>(ctx, wob, bo, out);
}
